// EMAMixer_33629593928093
// MI455X (gfx1250) — matrix operand order repaired and checked by reading, not compiled, not run on hardware
//
#include <hip/hip_runtime.h>
#include <hip/hip_bf16.h>

typedef __bf16 v8bf  __attribute__((ext_vector_type(8)));
typedef __bf16 v16bf __attribute__((ext_vector_type(16)));
typedef float  v8f   __attribute__((ext_vector_type(8)));
typedef unsigned int v4u __attribute__((ext_vector_type(4)));
typedef int v8i __attribute__((ext_vector_type(8)));
typedef int v4i __attribute__((ext_vector_type(4)));

static constexpr int Bb   = 8;
static constexpr int Tt   = 2048;
static constexpr int Ee   = 1024;
static constexpr int Mm   = 4;
static constexpr int Kdim = Ee * Mm;   // 4096 (GEMM K)
static constexpr int Ndim = Ee;        // 1024 (GEMM N)
static constexpr int Rows = Bb * Tt;   // 16384 (GEMM M)

__device__ __forceinline__ unsigned short f2bf(float f) {
  unsigned int u = __builtin_bit_cast(unsigned int, f);
  u += 0x7fffu + ((u >> 16) & 1u);   // round-to-nearest-even
  return (unsigned short)(u >> 16);
}

// ---------------------------------------------------------------------------
// Kernel 1: fused EMA recurrence, fp32 in -> bf16 out, laid out as the GEMM
// A matrix: yb[(b*T+t)][e*4+m].  One thread per (b,e); 4 independent chains.
// ---------------------------------------------------------------------------
__global__ __launch_bounds__(256)
void ema_kernel(const float* __restrict__ x,
                const float* __restrict__ alpha_logits,
                const float* __restrict__ bias,
                unsigned short* __restrict__ yb) {
  int gid = blockIdx.x * blockDim.x + threadIdx.x;   // 0 .. B*E-1
  if (gid >= Bb * Ee) return;
  int b = gid / Ee;
  int e = gid % Ee;

  float a[Mm], s[Mm], bi[Mm], acc[Mm];
#pragma unroll
  for (int m = 0; m < Mm; ++m) {
    float lg = alpha_logits[m];
    float am = 1.0f / (1.0f + __expf(-lg));
    a[m] = am; s[m] = 1.0f - am; bi[m] = bias[m]; acc[m] = 0.0f;
  }

  const float* xp = x + (size_t)b * Tt * Ee + e;           // stride Ee over t
  unsigned short* yp = yb + (size_t)b * Tt * Kdim + (size_t)e * Mm;

  for (int t = 0; t < Tt; ++t) {
    float xv = xp[(size_t)t * Ee];
    unsigned short h[Mm];
#pragma unroll
    for (int m = 0; m < Mm; ++m) {
      acc[m] = a[m] * acc[m] + (s[m] * xv + bi[m]);  // EMA + bias recurrence
      h[m] = f2bf(acc[m]);
    }
    uint2 pack;
    pack.x = (unsigned)h[0] | ((unsigned)h[1] << 16);
    pack.y = (unsigned)h[2] | ((unsigned)h[3] << 16);
    *(uint2*)(yp + (size_t)t * Kdim) = pack;               // 8B coalesced store
  }
}

// ---------------------------------------------------------------------------
// Kernel 2: W (K x N, f32) -> Wt (N x K, bf16), LDS-tiled transpose so both
// the read and the bf16 write are coalesced.
// ---------------------------------------------------------------------------
__global__ __launch_bounds__(256)
void wcast_kernel(const float* __restrict__ W, unsigned short* __restrict__ Wt) {
  __shared__ float tile[32][33];
  int n0 = blockIdx.x * 32;   // over N
  int k0 = blockIdx.y * 32;   // over K
  int tx = threadIdx.x;       // 0..31
  int ty = threadIdx.y;       // 0..7
#pragma unroll
  for (int i = 0; i < 32; i += 8)
    tile[ty + i][tx] = W[(size_t)(k0 + ty + i) * Ndim + (n0 + tx)];
  __syncthreads();
#pragma unroll
  for (int i = 0; i < 32; i += 8)
    Wt[(size_t)(n0 + ty + i) * Kdim + (k0 + tx)] = f2bf(tile[tx][ty + i]);
}

// ---------------------------------------------------------------------------
// TDM: issue a 2D tensor tile load (128 rows x 32 bf16 cols) into LDS.
// D# per CDNA5 ISA ch.8: group0 = {count, lds_addr, global_addr, type=2},
// group1 = {data_size=2B, pad 16B every 64B row, tensor dims/strides, tile
// dims}.  Hardware padding gives the LDS tile an 80B row pitch -> the 16-lane
// fragment reads advance 20 DWORD-banks per row (conflict-free mod 64).
// ---------------------------------------------------------------------------
#define LDS_PITCH 40   // shorts per tile row in LDS (64B data + 16B pad)

__device__ __forceinline__ void tdm_load_tile(unsigned lds_addr,
                                              const unsigned short* gptr,
                                              unsigned tensor_rows) {
  unsigned long long ga = (unsigned long long)(const void*)gptr;
  v4u g0;
  g0[0] = 1u;                                            // count=1 (valid D#)
  g0[1] = lds_addr;                                      // LDS byte address
  g0[2] = (unsigned)(ga & 0xffffffffull);                // global_addr[31:0]
  g0[3] = (unsigned)((ga >> 32) & 0x01ffffffull)         // global_addr[56:32]
        | (2u << 30);                                    // type=2 ("image")
  v8i g1;
  g1[0] = (1 << 16)      // data_size = 1 -> 2 bytes
        | (1 << 20)      // pad_enable
        | (3 << 22)      // pad_interval = 16 DWORDs (64B)
        | (3 << 25);     // pad_amount   = 4 DWORDs (16B)
  g1[1] = (int)((unsigned)(Kdim & 0xffff) << 16);        // tensor_dim0 lo16
  g1[2] = (int)((tensor_rows & 0xffffu) << 16);          // dim0 hi=0 | dim1 lo
  g1[3] = (int)((tensor_rows >> 16) & 0xffffu)           // tensor_dim1 hi16
        | (32 << 16);                                    // tile_dim0 = 32
  g1[4] = 128;                                           // tile_dim1=128, tile_dim2=0
  g1[5] = Kdim;                                          // tensor_dim0_stride lo32
  g1[6] = 0;                                             // stride hi | dim1_stride lo
  g1[7] = 0;
  v4i z4 = {0, 0, 0, 0};                                 // groups 2/3 unused (2D)
  v8i z8 = {0, 0, 0, 0, 0, 0, 0, 0};                     // extended slot unused
  __builtin_amdgcn_tensor_load_to_lds(g0, g1, z4, z4, z8, 0);
}

// ---------------------------------------------------------------------------
// Kernel 3: C[16384x1024] = A[16384x4096]bf16 * Wt^T + bias, fp32 accum.
// Block = 8 waves = 128x128 tile; per wave 64x32.  A/B tiles staged into LDS
// by the Tensor Data Mover (double-buffered, TENSORcnt-synchronized); waves
// consume fragments via ds_load and issue 8 v_wmma_f32_16x16x32_bf16/K-step.
// ---------------------------------------------------------------------------
__global__ __launch_bounds__(256)
void gemm_kernel(const unsigned short* __restrict__ A,
                 const unsigned short* __restrict__ Bt,   // Wt: N x K bf16
                 const float* __restrict__ bvec,
                 float* __restrict__ C) {
  // [ A buf0 | A buf1 | B buf0 | B buf1 ], each 128 rows * 40 shorts = 10240B
  __shared__ __align__(16) unsigned short lds[4 * 128 * LDS_PITCH];

  const int lane  = threadIdx.x & 31;
  const int wave  = threadIdx.x >> 5;
  const int lmod  = lane & 15;
  const int lhalf = lane >> 4;

  const int rowBlock = blockIdx.y * 128;
  const int colBlock = blockIdx.x * 128;
  const int aRow0 = (wave >> 2) * 64;          // wave's rows inside A tile
  const int bRow0 = (wave & 3) * 32;           // wave's cols inside B tile
  const int rowBase = rowBlock + aRow0;
  const int colBase = colBlock + bRow0;

  // Absolute LDS byte offsets of the four tile buffers (lds is the only
  // shared object; AS(3) cast yields the wave-relative LDS byte address).
  unsigned ldsBase =
      (unsigned)(unsigned long long)(__attribute__((address_space(3))) unsigned short*)&lds[0];
  const unsigned aBuf[2] = { ldsBase,             ldsBase + 10240u };
  const unsigned bBuf[2] = { ldsBase + 20480u,    ldsBase + 30720u };

  v8f acc[4][2];
#pragma unroll
  for (int i = 0; i < 4; ++i)
#pragma unroll
    for (int j = 0; j < 2; ++j)
      acc[i][j] = (v8f)(0.0f);

  const unsigned short* aG = A  + (size_t)rowBlock * Kdim;   // tile K origin moves
  const unsigned short* bG = Bt + (size_t)colBlock * Kdim;

  // Prologue: stage first K-slab into buffer 0.
  if (wave == 0) {
    tdm_load_tile(aBuf[0], aG, Rows);
    tdm_load_tile(bBuf[0], bG, Ndim);
  }

  for (int kb = 0; kb < Kdim; kb += 32) {
    const int buf = (kb >> 5) & 1;
    if (wave == 0) __builtin_amdgcn_s_wait_tensorcnt(0);
    __syncthreads();                          // tile visible to all waves
    if (wave == 0 && (kb + 32) < Kdim) {      // prefetch next slab into buf^1
      tdm_load_tile(aBuf[buf ^ 1], aG + kb + 32, Rows);
      tdm_load_tile(bBuf[buf ^ 1], bG + kb + 32, Ndim);
    }

    const unsigned short* aT = lds + (size_t)buf * 128 * LDS_PITCH;
    const unsigned short* bT = lds + (size_t)(2 + buf) * 128 * LDS_PITCH;

    // A fragment (16x32): lane = row lmod; K runs of 8 at lhalf*8 / 16+lhalf*8
    v16bf af[4];
#pragma unroll
    for (int i = 0; i < 4; ++i) {
      const unsigned short* r = aT + (size_t)(aRow0 + i * 16 + lmod) * LDS_PITCH;
      v8bf lo = *(const v8bf*)(r + lhalf * 8);
      v8bf hi = *(const v8bf*)(r + 16 + lhalf * 8);
      af[i] = __builtin_shufflevector(lo, hi, 0, 1, 2, 3, 4, 5, 6, 7,
                                      8, 9, 10, 11, 12, 13, 14, 15);
    }
    // B fragment (32x16): lane = col lmod; K runs of 8 at lhalf*8 / 16+lhalf*8
    v16bf bf[2];
#pragma unroll
    for (int j = 0; j < 2; ++j) {
      const unsigned short* r = bT + (size_t)(bRow0 + j * 16 + lmod) * LDS_PITCH
                                   + lhalf * 8;
      v8bf lo = *(const v8bf*)(r);
      v8bf hi = *(const v8bf*)(r + 16);
      bf[j] = __builtin_shufflevector(lo, hi, 0, 1, 2, 3, 4, 5, 6, 7,
                                      8, 9, 10, 11, 12, 13, 14, 15);
    }

#pragma unroll
    for (int i = 0; i < 4; ++i)
#pragma unroll
      for (int j = 0; j < 2; ++j)
        acc[i][j] = __builtin_amdgcn_wmma_f32_16x16x32_bf16(
            false, af[i], false, bf[j], (short)0, acc[i][j], false, false);

    __syncthreads();                          // done reading this buffer
  }

  // C/D layout: VGPR v -> row v + 8*lhalf; lane%16 -> column. Fuse bias.
#pragma unroll
  for (int j = 0; j < 2; ++j) {
    int col = colBase + j * 16 + lmod;
    float bb = bvec[col];
#pragma unroll
    for (int i = 0; i < 4; ++i) {
#pragma unroll
      for (int v = 0; v < 8; ++v) {
        int row = rowBase + i * 16 + v + lhalf * 8;
        C[(size_t)row * Ndim + col] = acc[i][j][v] + bb;
      }
    }
  }
}

// ---------------------------------------------------------------------------
extern "C" void kernel_launch(void* const* d_in, const int* in_sizes, int n_in,
                              void* d_out, int out_size, void* d_ws, size_t ws_size,
                              hipStream_t stream) {
  const float* x  = (const float*)d_in[0];   // (B,T,E)
  const float* al = (const float*)d_in[1];   // (M,)
  const float* bi = (const float*)d_in[2];   // (M,)
  const float* W  = (const float*)d_in[3];   // (E*M, E)
  const float* bv = (const float*)d_in[4];   // (E,)
  float* out = (float*)d_out;                // (B,T,E)

  unsigned short* yb = (unsigned short*)d_ws;                       // 128 MB
  unsigned short* Wt = (unsigned short*)((char*)d_ws +
                        (size_t)Rows * Kdim * sizeof(unsigned short)); // +8 MB

  // Phase 1: EMA -> bf16 A matrix
  ema_kernel<<<(Bb * Ee) / 256, 256, 0, stream>>>(x, al, bi, yb);

  // Phase 2: W -> Wt (transposed bf16)
  dim3 tb(32, 8);
  dim3 tg(Ndim / 32, Kdim / 32);
  wcast_kernel<<<tg, tb, 0, stream>>>(W, Wt);

  // Phase 3: TDM-staged WMMA GEMM + bias
  dim3 gg(Ndim / 128, Rows / 128);           // (8, 128) blocks of 8 waves
  gemm_kernel<<<gg, 256, 0, stream>>>(yb, Wt, bv, out);
}
